// LatticeLSTM_22995254903545
// MI455X (gfx1250) — matrix operand order repaired and checked by reading, not compiled, not run on hardware
//
#include <hip/hip_runtime.h>

// Problem dims (fixed by the reference)
#define T_LEN 4096
#define D_IN  128
#define H_DIM 512
#define H3    1536
#define E_DIM 256
#define WNUM  4
#define NBLK  12   // 12 blocks * 8 waves = 96 waves = 96 N-tiles of 16 over 3H=1536

typedef __attribute__((ext_vector_type(16))) __bf16 v16bf;
typedef __attribute__((ext_vector_type(8)))  __bf16 v8bf;
typedef __attribute__((ext_vector_type(8)))  float  v8f;

__device__ __forceinline__ float sigf(float x) { return 1.0f / (1.0f + __expf(-x)); }

__device__ __forceinline__ v16bf merge16(v8bf lo, v8bf hi) {
  return __builtin_shufflevector(lo, hi, 0, 1, 2, 3, 4, 5, 6, 7,
                                 8, 9, 10, 11, 12, 13, 14, 15);
}

// ---------------------------------------------------------------------------
// Weight swizzle: f32 row-major B[K,N] -> bf16 WMMA-fragment-ordered.
// For N-tile nt and K-chunk kc (32 rows), lane holds 16 contiguous bf16:
//   dst[(((nt*kcTotal)+kcBase+kc)*32 + lane)*16 + i] = B[kc*32+kb+(i<8?i:i+8)][nt*16+(lane&15)]
// with kb = (lane<16)?0:8.  One wave per (nt, kc).
// ---------------------------------------------------------------------------
__global__ void swizzle_weights(const float* __restrict__ src, __bf16* __restrict__ dst,
                                int N, int kcTotal, int kcBase) {
  const int nt = blockIdx.x;
  const int kc = blockIdx.y;
  const int lane = threadIdx.x & 31;
  const int kb = (lane < 16) ? 0 : 8;
  const int n = nt * 16 + (lane & 15);
  const float* sp = src + (size_t)(kc * 32 + kb) * N + n;
  __bf16* dp = dst + ((size_t)(nt * kcTotal + kcBase + kc) * 32 + lane) * 16;
#pragma unroll
  for (int i = 0; i < 16; ++i) dp[i] = (__bf16)sp[(size_t)(i < 8 ? i : i + 8) * N];
}

// ---------------------------------------------------------------------------
// Batched WMMA GEMM:  Out[M,N] = gather(A)[M,K] @ Bs[K,N](swizzled) + bias[N]
// One wave per 16x16 output tile.
// ---------------------------------------------------------------------------
template <typename OutT>
__global__ void wmma_gemm_bias(const float* __restrict__ A,
                               const int* __restrict__ rowidx,
                               const __bf16* __restrict__ Bs,
                               const float* __restrict__ bias,
                               OutT* __restrict__ Out, int N, int K) {
  const int lane = threadIdx.x & 31;
  const int m0 = blockIdx.y << 4;
  const int n0 = blockIdx.x << 4;
  const int mrow = lane & 15;
  const int kbA = (lane < 16) ? 0 : 8;
  const int n = n0 + (lane & 15);
  const int kcTotal = K >> 5;

  int m = m0 + mrow;
  long r = rowidx ? (long)rowidx[m] : (long)m;
  const float* Arow = A + (size_t)r * K;
  const __bf16* bp = Bs + ((size_t)(blockIdx.x * kcTotal) * 32 + lane) * 16;

  v8f c = {};
  for (int k0 = 0; k0 < K; k0 += 32) {
    // A fragment: two 32B vector loads + convert
    v8f alo = *(const v8f*)&Arow[k0 + kbA];
    v8f ahi = *(const v8f*)&Arow[k0 + 16 + kbA];
    v16bf a;
#pragma unroll
    for (int i = 0; i < 8; ++i) { a[i] = (__bf16)alo[i]; a[8 + i] = (__bf16)ahi[i]; }
    // B fragment: one 32B vector load (fragment-ordered)
    v16bf b = *(const v16bf*)bp;
    bp += 32 * 16;  // next K-chunk for this tile
    c = __builtin_amdgcn_wmma_f32_16x16x32_bf16(false, a, false, b,
                                                (short)0, c, false, false);
  }
  float bv = bias[n];
#pragma unroll
  for (int rr = 0; rr < 8; ++rr) {
    int mm = m0 + ((lane < 16) ? rr : 8 + rr);
    Out[(size_t)mm * N + n] = (OutT)(c[rr] + bv);
  }
}

// ---------------------------------------------------------------------------
// Grid-wide release/acquire barrier (persistent-kernel step sync).
// ---------------------------------------------------------------------------
__device__ __forceinline__ void grid_barrier(unsigned* bar, unsigned nb) {
  __threadfence();
  __syncthreads();
  if (threadIdx.x == 0) {
    unsigned gen = __hip_atomic_load(&bar[1], __ATOMIC_RELAXED, __HIP_MEMORY_SCOPE_AGENT);
    unsigned arrived =
        __hip_atomic_fetch_add(&bar[0], 1u, __ATOMIC_ACQ_REL, __HIP_MEMORY_SCOPE_AGENT);
    if (arrived == nb - 1u) {
      __hip_atomic_store(&bar[0], 0u, __ATOMIC_RELAXED, __HIP_MEMORY_SCOPE_AGENT);
      __hip_atomic_fetch_add(&bar[1], 1u, __ATOMIC_RELEASE, __HIP_MEMORY_SCOPE_AGENT);
    } else {
      while (__hip_atomic_load(&bar[1], __ATOMIC_ACQUIRE, __HIP_MEMORY_SCOPE_AGENT) == gen) {
        __builtin_amdgcn_s_sleep(1);
      }
    }
  }
  __syncthreads();
  __threadfence();
}

// ---------------------------------------------------------------------------
// Sequential lattice recurrence. 12 persistent blocks x 256 threads.
//  Phase A (all blocks): A1[16x1024] LDS, row0=[h_prev|0], rows1-4=[0|h_s_w];
//    96 waves each WMMA one 16-col tile vs Bcomb=[U;Uw] (swizzled bf16);
//    add precomputed x/emb projections; rows0-4 -> Gbuf.  -> grid barrier
//  Phase B (block 0): c_w elementwise; c_w@Ua via WMMA; alpha; softmax cell
//    mix; h1/c1 -> out row t.  -> grid barrier
// d_out doubles as the Hbuf/Cbuf lattice state: out[t] = [h(512) | c(512)].
// ---------------------------------------------------------------------------
__global__ void __launch_bounds__(256, 1)
lattice_recurrent(const int* __restrict__ wstarts,
                  const unsigned char* __restrict__ wmask,
                  const float* __restrict__ h0, const float* __restrict__ c0,
                  const __bf16* __restrict__ BcombS,  // swizzled [K=1024,N=1536]
                  const __bf16* __restrict__ UaS,     // swizzled [K=512,N=512]
                  const float* __restrict__ XW,       // [T][1536]  x@W + b
                  const float* __restrict__ XWa,      // [T][512]   x@Wa + ba
                  const __bf16* __restrict__ XE,      // [T*4][1536] emb@Ww + bw
                  float* __restrict__ Gbuf,           // [5][1536]
                  unsigned* __restrict__ bar,         // [2]
                  float* __restrict__ out)            // [T][1024]
{
  __shared__ __bf16 A1[16][1024];   // 32 KB: stacked-K A matrix
  __shared__ float  cwS[WNUM][H_DIM];
  __shared__ __bf16 A2[WNUM][H_DIM];
  __shared__ float  alS[WNUM][H_DIM];

  const int tid  = threadIdx.x;
  const int lane = tid & 31;
  const int wave = tid >> 5;
  const int blk  = blockIdx.x;
  const int mrow = lane & 15;
  const int kbA  = (lane < 16) ? 0 : 8;

  // Zero A1 once; per step only the live segments are rewritten.
  for (int i = tid; i < 16 * 1024; i += 256) (&A1[0][0])[i] = (__bf16)0.0f;
  __syncthreads();

  const int nt = blk * 8 + wave;  // 0..95, this wave's N-tile of 1536
  const __bf16* bbase = BcombS + ((size_t)(nt * 32) * 32 + lane) * 16;

  for (int t = 0; t < T_LEN; ++t) {
    // ---- Phase A: build A1 and run the fused [U;Uw] WMMA pass -------------
    for (int j = tid; j < H_DIM; j += 256) {
      float hp = (t == 0) ? h0[j] : out[(size_t)(t - 1) * 1024 + j];
      A1[0][j] = (__bf16)hp;
    }
    for (int j = tid; j < WNUM * H_DIM; j += 256) {
      int w = j >> 9, jj = j & 511;
      float hv = 0.0f;
      if (wmask[t * WNUM + w]) {
        int s = wstarts[t * WNUM + w];  // s <= t-1 when masked-on
        hv = out[(size_t)s * 1024 + jj];
      }
      A1[1 + w][H_DIM + jj] = (__bf16)hv;
    }
    __syncthreads();

    const int n = nt * 16 + (lane & 15);
    const __bf16* bp = bbase;
    v8f c = {};
    for (int k0 = 0; k0 < 1024; k0 += 32) {
      v8bf lo = *(const v8bf*)&A1[mrow][k0 + kbA];
      v8bf hi = *(const v8bf*)&A1[mrow][k0 + 16 + kbA];
      v16bf a = merge16(lo, hi);
      v16bf b = *(const v16bf*)bp;
      __builtin_prefetch(bp + 2 * 32 * 16, 0, 1);  // global_prefetch: K+2 chunk
      bp += 32 * 16;
      c = __builtin_amdgcn_wmma_f32_16x16x32_bf16(false, a, false, b,
                                                  (short)0, c, false, false);
    }
    if (lane < 16) {
      Gbuf[n] = c[0] + XW[(size_t)t * H3 + n];  // gates row
#pragma unroll
      for (int rr = 1; rr < 5; ++rr)            // gw rows
        Gbuf[rr * H3 + n] =
            c[rr] + (float)XE[((size_t)t * WNUM + (rr - 1)) * H3 + n];
    }
    grid_barrier(bar, (unsigned)gridDim.x);

    // ---- Phase B: block 0 does word cells, alpha WMMA, and the combine ----
    if (blk == 0) {
      for (int j = tid; j < WNUM * H_DIM; j += 256) {
        int w = j >> 9, jj = j & 511;
        float cw = 0.0f;
        if (wmask[t * WNUM + w]) {
          int s = wstarts[t * WNUM + w];
          float cs = out[(size_t)s * 1024 + H_DIM + jj];
          float fw = sigf(Gbuf[(1 + w) * H3 + jj]);
          float iw = sigf(Gbuf[(1 + w) * H3 + H_DIM + jj]);
          float gw = tanhf(Gbuf[(1 + w) * H3 + 2 * H_DIM + jj]);
          cw = fw * cs + iw * gw;
        }
        cwS[w][jj] = cw;
        A2[w][jj] = (__bf16)cw;
      }
      __syncthreads();

      // alpha_lin[4,512] = c_w @ Ua + XWa[t]; 8 waves x 4 N-tiles
      for (int tt = 0; tt < 4; ++tt) {
        const int nta = wave * 4 + tt;  // 0..31
        const int na = nta * 16 + (lane & 15);
        const __bf16* bp2 = UaS + ((size_t)(nta * 16) * 32 + lane) * 16;
        v8f ca = {};
        for (int k0 = 0; k0 < H_DIM; k0 += 32) {
          v16bf a = {};
          if (mrow < WNUM) {
            v8bf lo = *(const v8bf*)&A2[mrow][k0 + kbA];
            v8bf hi = *(const v8bf*)&A2[mrow][k0 + 16 + kbA];
            a = merge16(lo, hi);
          }
          v16bf b = *(const v16bf*)bp2;
          bp2 += 32 * 16;
          ca = __builtin_amdgcn_wmma_f32_16x16x32_bf16(false, a, false, b,
                                                       (short)0, ca, false, false);
        }
        if (lane < 16) {
#pragma unroll
          for (int rr = 0; rr < WNUM; ++rr)
            alS[rr][na] = ca[rr] + XWa[(size_t)t * H_DIM + na];
        }
      }
      __syncthreads();

      const bool many = wmask[t * WNUM] || wmask[t * WNUM + 1] ||
                        wmask[t * WNUM + 2] || wmask[t * WNUM + 3];
      for (int jj = tid; jj < H_DIM; jj += 256) {
        float ig = sigf(Gbuf[jj]);
        float og = sigf(Gbuf[H_DIM + jj]);
        float gg = tanhf(Gbuf[2 * H_DIM + jj]);
        float cprev = (t == 0) ? c0[jj] : out[(size_t)(t - 1) * 1024 + H_DIM + jj];
        float e0 = __expf(ig);
        float ssum = e0, acc = e0 * gg;
#pragma unroll
        for (int w = 0; w < WNUM; ++w) {
          if (wmask[t * WNUM + w]) {
            float aw = __expf(sigf(alS[w][jj]));
            ssum += aw;
            acc += aw * cwS[w][jj];
          }
        }
        float c1 = many ? (acc / ssum) : ((1.0f - ig) * cprev + ig * gg);
        float h1 = og * tanhf(c1);
        out[(size_t)t * 1024 + jj] = h1;
        out[(size_t)t * 1024 + H_DIM + jj] = c1;
      }
    }
    grid_barrier(bar, (unsigned)gridDim.x);
  }
}

// ---------------------------------------------------------------------------
// Host launcher
// ---------------------------------------------------------------------------
extern "C" void kernel_launch(void* const* d_in, const int* in_sizes, int n_in,
                              void* d_out, int out_size, void* d_ws, size_t ws_size,
                              hipStream_t stream) {
  (void)in_sizes; (void)n_in; (void)out_size; (void)ws_size;
  const float* x        = (const float*)d_in[0];
  const int*   word_ids = (const int*)d_in[1];
  const int*   wstarts  = (const int*)d_in[2];
  const unsigned char* wmask = (const unsigned char*)d_in[3];
  const float* h0 = (const float*)d_in[4];
  const float* c0 = (const float*)d_in[5];
  const float* emb = (const float*)d_in[6];
  const float* W  = (const float*)d_in[7];
  const float* U  = (const float*)d_in[8];
  const float* b  = (const float*)d_in[9];
  const float* Wa = (const float*)d_in[10];
  const float* Ua = (const float*)d_in[11];
  const float* ba = (const float*)d_in[12];
  const float* Ww = (const float*)d_in[13];
  const float* Uw = (const float*)d_in[14];
  const float* bw = (const float*)d_in[15];
  float* out = (float*)d_out;

  char* ws = (char*)d_ws;
  size_t off = 0;
  auto take = [&](size_t bytes) {
    size_t o = off; off = (off + bytes + 255) & ~(size_t)255; return o;
  };
  unsigned* bar    = (unsigned*)(ws + take(256));
  __bf16*   BcombS = (__bf16*)(ws + take((size_t)1024 * H3 * 2));   // [U;Uw] swizzled
  __bf16*   UaS    = (__bf16*)(ws + take((size_t)H_DIM * H_DIM * 2));
  __bf16*   WS     = (__bf16*)(ws + take((size_t)D_IN * H3 * 2));
  __bf16*   WaS    = (__bf16*)(ws + take((size_t)D_IN * H_DIM * 2));
  __bf16*   WwS    = (__bf16*)(ws + take((size_t)E_DIM * H3 * 2));
  float*    XW     = (float*)(ws + take((size_t)T_LEN * H3 * 4));
  float*    XWa    = (float*)(ws + take((size_t)T_LEN * H_DIM * 4));
  __bf16*   XE     = (__bf16*)(ws + take((size_t)T_LEN * WNUM * H3 * 2));
  float*    Gbuf   = (float*)(ws + take((size_t)5 * H3 * 4));

  hipMemsetAsync(bar, 0, 256, stream);

  // Swizzle+convert all B-side weights into WMMA fragment order.
  swizzle_weights<<<dim3(H3 / 16, 16),    32, 0, stream>>>(U,  BcombS, H3,    32, 0);
  swizzle_weights<<<dim3(H3 / 16, 16),    32, 0, stream>>>(Uw, BcombS, H3,    32, 16);
  swizzle_weights<<<dim3(H_DIM / 16, 16), 32, 0, stream>>>(Ua, UaS,    H_DIM, 16, 0);
  swizzle_weights<<<dim3(H3 / 16, 4),     32, 0, stream>>>(W,  WS,     H3,    4,  0);
  swizzle_weights<<<dim3(H_DIM / 16, 4),  32, 0, stream>>>(Wa, WaS,    H_DIM, 4,  0);
  swizzle_weights<<<dim3(H3 / 16, 8),     32, 0, stream>>>(Ww, WwS,    H3,    8,  0);

  // Parallel pre-projections (bulk of the FLOPs, fully WMMA):
  wmma_gemm_bias<float><<<dim3(H3 / 16, T_LEN / 16), 32, 0, stream>>>(
      x, nullptr, WS, b, XW, H3, D_IN);
  wmma_gemm_bias<float><<<dim3(H_DIM / 16, T_LEN / 16), 32, 0, stream>>>(
      x, nullptr, WaS, ba, XWa, H_DIM, D_IN);
  wmma_gemm_bias<__bf16><<<dim3(H3 / 16, (T_LEN * WNUM) / 16), 32, 0, stream>>>(
      emb, word_ids, WwS, bw, XE, H3, E_DIM);

  // Sequential lattice recurrence (persistent cooperative kernel)
  lattice_recurrent<<<NBLK, 256, 0, stream>>>(
      wstarts, wmask, h0, c0, BcombS, UaS, XW, XWa, XE, Gbuf, bar, out);
}
